// SpikePointMamba_8744553414799
// MI455X (gfx1250) — matrix operand order repaired and checked by reading, not compiled, not run on hardware
//
#include <hip/hip_runtime.h>
#include <hip/hip_fp16.h>

// ---------------------------------------------------------------------------
// SpikePointMamba forward for MI455X (gfx1250, wave32, WMMA).
// GEMMs: v_wmma_f32_16x16x32_f16, 64x64 block tile, 4 waves, 2x2 WMMA tiles
// per wave with fragment reuse; async global->LDS staging (ASYNCcnt path).
// Weights pre-converted once to f16 [N][K]. BN = two-pass atomic stats.
// ---------------------------------------------------------------------------

typedef __attribute__((ext_vector_type(16))) _Float16 v16h;
typedef __attribute__((ext_vector_type(8)))  _Float16 h8;
typedef __attribute__((ext_vector_type(8)))  float    v8f;
typedef __attribute__((ext_vector_type(4)))  int      v4i;

#if __has_builtin(__builtin_amdgcn_global_load_async_to_lds_b128) && \
    __has_builtin(__builtin_amdgcn_s_wait_asynccnt)
#define USE_ASYNC_LDS 1
#else
#define USE_ASYNC_LDS 0
#endif

#define T_    2
#define B_    8
#define N_    1024
#define G_    256
#define K_    32
#define D_    384
#define DEPTH_ 12
#define DI_   768
#define DS_   16
#define DC_   4
#define DTR_  24
#define ENC_  384
#define CLS_  40
#define TB_   16
#define MTOK  (TB_*G_)          /* 4096 group tokens  */
#define MPT   (TB_*G_*K_)       /* 131072 point tokens */

#define CDIV(a,b) (((a)+(b)-1)/(b))

__device__ __forceinline__ v8f wmma16(v16h a, v16h b, v8f c) {
  return __builtin_amdgcn_wmma_f32_16x16x32_f16(false, a, false, b,
                                                (short)0, c, false, false);
}

// ---------------------------------------------------------------------------
// Weight prep: f32 [K,N] (trans=1) or f32 [N,K] (trans=0) -> f16 [N][ldw],
// zero-padded K tail so GEMM rows are 16B-vector addressable.
// ---------------------------------------------------------------------------
__global__ __launch_bounds__(256)
void k_wprep(const float* __restrict__ W, _Float16* __restrict__ Wt,
             int Kd, int Nn, int ldw, int trans)
{
  size_t i = (size_t)blockIdx.x * blockDim.x + threadIdx.x;
  if (i >= (size_t)Nn * ldw) return;
  int k = (int)(i % ldw);
  size_t n = i / ldw;
  float v = 0.0f;
  if (k < Kd) v = trans ? W[(size_t)k * Nn + n] : W[n * (size_t)Kd + k];
  Wt[i] = (_Float16)v;
}

// ---------------------------------------------------------------------------
// WMMA GEMM: C[M,N](f16) = A[M,Kd](f16) * Bt[N,Kd](f16) + bias.
// 128 threads = 4 waves; block tile 64x64; wave quadrant 32x32 = 2x2 WMMA.
// ---------------------------------------------------------------------------
__global__ __launch_bounds__(128)
void k_gemm(const _Float16* __restrict__ A, int lda,
            const _Float16* __restrict__ Bw, int ldb,
            const float* __restrict__ bias,
            _Float16* __restrict__ C, int ldc,
            int M, int Nn, int Kd)
{
  __shared__ __align__(16) _Float16 As[64][48];   // [m][k], 96B row stride
  __shared__ __align__(16) _Float16 Bs[64][48];   // [n][k]

  const int tid  = threadIdx.x;
  const int lane = tid & 31;
  const int wave = tid >> 5;
  const int qm = (wave >> 1) * 32, qn = (wave & 1) * 32;
  const int bm = blockIdx.y * 64, bn = blockIdx.x * 64;
  const int srow = tid >> 1;        // 0..63
  const int sseg = (tid & 1) * 16;  // 0 or 16

  const bool avec = ((lda & 7) == 0);
  const bool bvec = ((ldb & 7) == 0);

  v8f acc[2][2] = {};

  for (int k0 = 0; k0 < Kd; k0 += 32) {
    // ---- stage A tile ----
    {
      int gm = bm + srow;
      const _Float16* src = A + (size_t)gm * lda + k0 + sseg;
      if (avec && gm < M && (k0 + 32) <= Kd) {
#if USE_ASYNC_LDS
        __builtin_amdgcn_global_load_async_to_lds_b128(
            (v4i*)src, (v4i*)&As[srow][sseg], 0, 0);
        __builtin_amdgcn_global_load_async_to_lds_b128(
            (v4i*)(src + 8), (v4i*)&As[srow][sseg + 8], 0, 0);
#else
        *(h8*)&As[srow][sseg]     = *(const h8*)src;
        *(h8*)&As[srow][sseg + 8] = *(const h8*)(src + 8);
#endif
        if (k0 + 64 <= Kd) __builtin_prefetch(src + 32, 0, 1);
      } else {
        #pragma unroll
        for (int i = 0; i < 16; ++i) {
          int kk = k0 + sseg + i;
          _Float16 v = (_Float16)0.0f;
          if (gm < M && kk < Kd) v = A[(size_t)gm * lda + kk];
          As[srow][sseg + i] = v;
        }
      }
    }
    // ---- stage B tile ----
    {
      int gn = bn + srow;
      const _Float16* src = Bw + (size_t)gn * ldb + k0 + sseg;
      if (bvec && gn < Nn && (k0 + 32) <= Kd) {
#if USE_ASYNC_LDS
        __builtin_amdgcn_global_load_async_to_lds_b128(
            (v4i*)src, (v4i*)&Bs[srow][sseg], 0, 0);
        __builtin_amdgcn_global_load_async_to_lds_b128(
            (v4i*)(src + 8), (v4i*)&Bs[srow][sseg + 8], 0, 0);
#else
        *(h8*)&Bs[srow][sseg]     = *(const h8*)src;
        *(h8*)&Bs[srow][sseg + 8] = *(const h8*)(src + 8);
#endif
        if (k0 + 64 <= Kd) __builtin_prefetch(src + 32, 0, 1);
      } else {
        #pragma unroll
        for (int i = 0; i < 16; ++i) {
          int kk = k0 + sseg + i;
          _Float16 v = (_Float16)0.0f;
          if (gn < Nn && kk < Kd) v = Bw[(size_t)gn * ldb + kk];
          Bs[srow][sseg + i] = v;
        }
      }
    }
#if USE_ASYNC_LDS
    __builtin_amdgcn_s_wait_asynccnt(0);
#endif
    __syncthreads();

    // ---- fragments (ISA 16-bit layouts) ----
    // A 16x32: lanes 0-15 -> K{0..7,16..23}; lanes 16-31 -> K{8..15,24..31}
    // B 32x16: lanes 0-15 -> K{0..7,16..23}; lanes 16-31 -> K{8..15,24..31}
    const int r  = lane & 15;
    const int hi = lane >> 4;
    const int ka = hi * 8;
    const int kb = hi * 8;
    v16h af[2], bf[2];
    #pragma unroll
    for (int ti = 0; ti < 2; ++ti) {
      int arow = qm + ti * 16 + r;
      h8 lo = *(const h8*)&As[arow][ka];
      h8 hv = *(const h8*)&As[arow][16 + ka];
      #pragma unroll
      for (int i = 0; i < 8; ++i) { af[ti][i] = lo[i]; af[ti][8 + i] = hv[i]; }
    }
    #pragma unroll
    for (int tj = 0; tj < 2; ++tj) {
      int brow = qn + tj * 16 + r;
      h8 lo = *(const h8*)&Bs[brow][kb];
      h8 hv = *(const h8*)&Bs[brow][16 + kb];
      #pragma unroll
      for (int i = 0; i < 8; ++i) { bf[tj][i] = lo[i]; bf[tj][8 + i] = hv[i]; }
    }
    #pragma unroll
    for (int ti = 0; ti < 2; ++ti)
      #pragma unroll
      for (int tj = 0; tj < 2; ++tj)
        acc[ti][tj] = wmma16(af[ti], bf[tj], acc[ti][tj]);
    __syncthreads();
  }

  // ---- store: C VGPR i -> M = i (+8 for hi lanes), N = lane&15 ----
  const int r  = lane & 15;
  const int hi = lane >> 4;
  #pragma unroll
  for (int tj = 0; tj < 2; ++tj) {
    int gn = bn + qn + tj * 16 + r;
    if (gn >= Nn) continue;
    float bv = bias ? bias[gn] : 0.0f;
    #pragma unroll
    for (int ti = 0; ti < 2; ++ti) {
      #pragma unroll
      for (int i = 0; i < 8; ++i) {
        int gm = bm + qm + ti * 16 + hi * 8 + i;
        if (gm < M) C[(size_t)gm * ldc + gn] = (_Float16)(acc[ti][tj][i] + bv);
      }
    }
  }
}

// ---------------------------------------------------------------------------
// Furthest point sampling: one block per batch, 256 threads, dists in regs.
// ---------------------------------------------------------------------------
__global__ __launch_bounds__(256)
void k_fps(const float* __restrict__ pts, int* __restrict__ idx,
           float* __restrict__ cen0, _Float16* __restrict__ cenh)
{
  const int b   = blockIdx.x;
  const int tid = threadIdx.x;
  __shared__ float lp[3];
  __shared__ float sval[256];
  __shared__ int   sidx[256];
  __shared__ int   sel[G_];

  float dl[4];
  #pragma unroll
  for (int j = 0; j < 4; ++j) dl[j] = 1e10f;

  int last = 0;
  if (tid == 0) sel[0] = 0;
  __syncthreads();

  for (int i = 1; i < G_; ++i) {
    if (tid < 3) lp[tid] = pts[((size_t)b * N_ + last) * 3 + tid];
    __syncthreads();
    float bestv = -1.0f; int bestn = 0;
    #pragma unroll
    for (int j = 0; j < 4; ++j) {
      int n = tid + 256 * j;
      float dx = pts[((size_t)b * N_ + n) * 3 + 0] - lp[0];
      float dy = pts[((size_t)b * N_ + n) * 3 + 1] - lp[1];
      float dz = pts[((size_t)b * N_ + n) * 3 + 2] - lp[2];
      float d = dx * dx + dy * dy + dz * dz;
      if (d < dl[j]) dl[j] = d;
      if (dl[j] > bestv) { bestv = dl[j]; bestn = n; }
    }
    sval[tid] = bestv; sidx[tid] = bestn;
    __syncthreads();
    for (int off = 128; off > 0; off >>= 1) {
      if (tid < off) {
        if (sval[tid + off] > sval[tid] ||
            (sval[tid + off] == sval[tid] && sidx[tid + off] < sidx[tid])) {
          sval[tid] = sval[tid + off]; sidx[tid] = sidx[tid + off];
        }
      }
      __syncthreads();
    }
    last = sidx[0];
    if (tid == 0) sel[i] = last;
    __syncthreads();
  }

  for (int g = tid; g < G_; g += 256) {
    int gi = sel[g];
    idx[b * G_ + g] = gi;
    float p[3];
    #pragma unroll
    for (int c = 0; c < 3; ++c) {
      p[c] = pts[((size_t)b * N_ + gi) * 3 + c];
      cen0[((size_t)b * G_ + g) * 3 + c] = p[c];
    }
    #pragma unroll
    for (int t = 0; t < T_; ++t) {
      size_t tok = ((size_t)(t * B_ + b) * G_ + g) * 8;
      #pragma unroll
      for (int c = 0; c < 8; ++c)
        cenh[tok + c] = (_Float16)(c < 3 ? p[c] : 0.0f);
    }
  }
}

// ---------------------------------------------------------------------------
// KNN (K=32 nearest of N=1024) + gather (center-subtracted, f16, channel-
// padded to 8, duplicated over T). One 128-thread block per (g,b).
// ---------------------------------------------------------------------------
__global__ __launch_bounds__(128)
void k_knn(const float* __restrict__ pts, const float* __restrict__ cen0,
           _Float16* __restrict__ nb)
{
  const int g = blockIdx.x, b = blockIdx.y, tid = threadIdx.x;
  __shared__ float d2[N_];
  __shared__ float mval[128];
  __shared__ int   midx[128];

  float cx = cen0[((size_t)b * G_ + g) * 3 + 0];
  float cy = cen0[((size_t)b * G_ + g) * 3 + 1];
  float cz = cen0[((size_t)b * G_ + g) * 3 + 2];

  #pragma unroll
  for (int j = 0; j < 8; ++j) {
    int n = tid + 128 * j;
    float dx = pts[((size_t)b * N_ + n) * 3 + 0] - cx;
    float dy = pts[((size_t)b * N_ + n) * 3 + 1] - cy;
    float dz = pts[((size_t)b * N_ + n) * 3 + 2] - cz;
    d2[n] = dx * dx + dy * dy + dz * dz;
  }
  __syncthreads();

  for (int kk = 0; kk < K_; ++kk) {
    float best = 3.0e38f; int bi = 0;
    #pragma unroll
    for (int j = 0; j < 8; ++j) {
      int n = tid + 128 * j;
      float v = d2[n];
      if (v < best || (v == best && n < bi)) { best = v; bi = n; }
    }
    mval[tid] = best; midx[tid] = bi;
    __syncthreads();
    for (int off = 64; off > 0; off >>= 1) {
      if (tid < off) {
        if (mval[tid + off] < mval[tid] ||
            (mval[tid + off] == mval[tid] && midx[tid + off] < midx[tid])) {
          mval[tid] = mval[tid + off]; midx[tid] = midx[tid + off];
        }
      }
      __syncthreads();
    }
    int ch = midx[0];
    if (tid == 0) d2[ch] = 3.0e38f;
    if (tid < 16) {
      int t = tid >> 3, c = tid & 7;
      float v = 0.0f;
      if (c == 0) v = pts[((size_t)b * N_ + ch) * 3 + 0] - cx;
      if (c == 1) v = pts[((size_t)b * N_ + ch) * 3 + 1] - cy;
      if (c == 2) v = pts[((size_t)b * N_ + ch) * 3 + 2] - cz;
      size_t tok = (((size_t)(t * B_ + b) * G_ + g) * K_ + kk) * 8;
      nb[tok + c] = (_Float16)v;
    }
    __syncthreads();
  }
}

// ---------------------------------------------------------------------------
// BN batch statistics: atomic per-channel sum / sumsq over dense [M,C] f16.
// ---------------------------------------------------------------------------
__global__ __launch_bounds__(256)
void k_stats(const _Float16* __restrict__ X, int C, int M,
             float* __restrict__ gsum, float* __restrict__ gss)
{
  int c = blockIdx.x * blockDim.x + threadIdx.x;
  if (c >= C) return;
  float s = 0.0f, ss = 0.0f;
  for (int m = blockIdx.y; m < M; m += gridDim.y) {
    float v = (float)X[(size_t)m * C + c];
    s += v; ss += v * v;
  }
  atomicAdd(&gsum[c], s);
  atomicAdd(&gss[c], ss);
}

// BN apply (no LIF), dense src [M,C] -> dst [m*ldy+yoff+c]
__global__ __launch_bounds__(256)
void k_bn_apply(const _Float16* __restrict__ X, int C, int M,
                _Float16* __restrict__ Y, int ldy, int yoff,
                const float* __restrict__ gsum, const float* __restrict__ gss,
                const float* __restrict__ gam, const float* __restrict__ bet,
                float cnt)
{
  size_t i = (size_t)blockIdx.x * blockDim.x + threadIdx.x;
  if (i >= (size_t)M * C) return;
  int c = (int)(i % C); size_t m = i / C;
  float mu = gsum[c] / cnt;
  float va = gss[c] / cnt - mu * mu;
  float v = ((float)X[i] - mu) * rsqrtf(va + 1e-5f) * gam[c] + bet[c];
  Y[m * ldy + yoff + c] = (_Float16)v;
}

// LIF over T=2 (rows [0,Mhalf) are t=0), optional fused BN first.
__global__ __launch_bounds__(256)
void k_lif(const _Float16* __restrict__ X, int ldx, int xoff,
           _Float16* __restrict__ Y, int ldy, int yoff,
           int C, int Mhalf,
           const float* __restrict__ gsum, const float* __restrict__ gss,
           const float* __restrict__ gam, const float* __restrict__ bet,
           float cnt)
{
  size_t i = (size_t)blockIdx.x * blockDim.x + threadIdx.x;
  if (i >= (size_t)Mhalf * C) return;
  int c = (int)(i % C); size_t m = i / C;
  float x1 = (float)X[m * ldx + xoff + c];
  float x2 = (float)X[(m + Mhalf) * ldx + xoff + c];
  if (gsum) {
    float mu = gsum[c] / cnt;
    float va = gss[c] / cnt - mu * mu;
    float is = rsqrtf(va + 1e-5f);
    x1 = (x1 - mu) * is * gam[c] + bet[c];
    x2 = (x2 - mu) * is * gam[c] + bet[c];
  }
  float v1 = x1 * 0.5f;                    // TAU = 0.5, v0 = 0
  float s1 = (v1 >= 1.0f) ? 1.0f : 0.0f;   // VTH = 1
  float vr = v1 * (1.0f - s1);             // hard reset
  float v2 = vr + (x2 - vr) * 0.5f;
  float s2 = (v2 >= 1.0f) ? 1.0f : 0.0f;
  Y[m * ldy + yoff + c] = (_Float16)s1;
  Y[(m + Mhalf) * ldy + yoff + c] = (_Float16)s2;
}

// max over inner K axis; bcast=1 replicates result to all K rows of dst
__global__ __launch_bounds__(256)
void k_maxk(const _Float16* __restrict__ X, int ldx, int xoff,
            _Float16* __restrict__ Y, int ldy, int yoff,
            int Gn, int Kn, int C, int bcast)
{
  size_t i = (size_t)blockIdx.x * blockDim.x + threadIdx.x;
  if (i >= (size_t)Gn * C) return;
  int c = (int)(i % C); size_t grp = i / C;
  float mv = -3.0e38f;
  for (int k = 0; k < Kn; ++k) {
    float v = (float)X[(grp * Kn + k) * ldx + xoff + c];
    mv = v > mv ? v : mv;
  }
  if (bcast) {
    for (int k = 0; k < Kn; ++k)
      Y[(grp * Kn + k) * ldy + yoff + c] = (_Float16)mv;
  } else {
    Y[grp * ldy + yoff + c] = (_Float16)mv;
  }
}

__global__ __launch_bounds__(256)
void k_add(const _Float16* __restrict__ A, const _Float16* __restrict__ Bp,
           _Float16* __restrict__ O, size_t n)
{
  size_t i = (size_t)blockIdx.x * blockDim.x + threadIdx.x;
  if (i < n) O[i] = (_Float16)((float)A[i] + (float)Bp[i]);
}

// Depthwise causal conv (DC=4, left pad 3) on the x-half of xz, + bias, LIF.
__global__ __launch_bounds__(256)
void k_conv_lif(const _Float16* __restrict__ xz,
                const float* __restrict__ cw, const float* __restrict__ cb,
                _Float16* __restrict__ sx)
{
  size_t i = (size_t)blockIdx.x * blockDim.x + threadIdx.x;
  if (i >= (size_t)B_ * G_ * DI_) return;
  int d = (int)(i % DI_);
  size_t r = i / DI_;
  int g = (int)(r % G_);
  int b = (int)(r / G_);
  float vt[2];
  #pragma unroll
  for (int t = 0; t < T_; ++t) {
    float acc = cb[d];
    #pragma unroll
    for (int j = 0; j < DC_; ++j) {
      int gg = g - 3 + j;
      if (gg >= 0)
        acc += cw[d * DC_ + j] *
               (float)xz[((size_t)((t * B_ + b) * G_ + gg)) * (2 * DI_) + d];
    }
    vt[t] = acc;
  }
  float v1 = vt[0] * 0.5f;
  float s1 = (v1 >= 1.0f) ? 1.0f : 0.0f;
  float vr = v1 * (1.0f - s1);
  float v2 = vr + (vt[1] - vr) * 0.5f;
  float s2 = (v2 >= 1.0f) ? 1.0f : 0.0f;
  sx[((size_t)((0 * B_ + b) * G_ + g)) * DI_ + d] = (_Float16)s1;
  sx[((size_t)((1 * B_ + b) * G_ + g)) * DI_ + d] = (_Float16)s2;
}

__global__ __launch_bounds__(256)
void k_softplus(const _Float16* __restrict__ raw, const float* __restrict__ bdt,
                float* __restrict__ dt, size_t n)
{
  size_t i = (size_t)blockIdx.x * blockDim.x + threadIdx.x;
  if (i >= n) return;
  int d = (int)(i % DI_);
  float v = (float)raw[i] + bdt[d];
  dt[i] = (v > 20.0f) ? v : log1pf(expf(v));
}

// Selective scan: one thread per (tb, d), DS=16 states in registers.
__global__ __launch_bounds__(256)
void k_scan(const float* __restrict__ dt, const _Float16* __restrict__ dbl,
            const _Float16* __restrict__ sx, const _Float16* __restrict__ sz,
            const float* __restrict__ Alog, const float* __restrict__ Dp,
            _Float16* __restrict__ ybuf)
{
  int i = blockIdx.x * blockDim.x + threadIdx.x;
  if (i >= TB_ * DI_) return;
  int d = i % DI_;
  int tb = i / DI_;
  float Av[DS_], hs[DS_];
  #pragma unroll
  for (int s = 0; s < DS_; ++s) {
    Av[s] = -expf(Alog[(size_t)d * DS_ + s]);
    hs[s] = 0.0f;
  }
  float dp = Dp[d];
  for (int g = 0; g < G_; ++g) {
    size_t base = (size_t)tb * G_ + g;
    float dtv = dt[base * DI_ + d];
    float xv  = (float)sx[base * DI_ + d];
    float dtx = dtv * xv;
    float y = 0.0f;
    #pragma unroll
    for (int s = 0; s < DS_; ++s) {
      float bm = (float)dbl[base * 56 + DTR_ + s];
      float cm = (float)dbl[base * 56 + DTR_ + DS_ + s];
      hs[s] = expf(dtv * Av[s]) * hs[s] + dtx * bm;
      y += hs[s] * cm;
    }
    y = (y + xv * dp) * (float)sz[base * DI_ + d];
    ybuf[base * DI_ + d] = (_Float16)y;
  }
}

__global__ __launch_bounds__(256)
void k_final(const _Float16* __restrict__ lg, float* __restrict__ out)
{
  int i = blockIdx.x * blockDim.x + threadIdx.x;
  if (i >= B_ * CLS_) return;
  int b = i / CLS_, c = i % CLS_;
  out[i] = 0.5f * ((float)lg[b * CLS_ + c] + (float)lg[(B_ + b) * CLS_ + c]);
}

// ---------------------------------------------------------------------------
// Host orchestration
// ---------------------------------------------------------------------------
static void gemm(hipStream_t s, const _Float16* A, int lda,
                 const _Float16* Bw, int ldb, const float* bias,
                 _Float16* C, int ldc, int M, int Nn, int Kd)
{
  dim3 g(CDIV(Nn, 64), CDIV(M, 64));
  k_gemm<<<g, dim3(128), 0, s>>>(A, lda, Bw, ldb, bias, C, ldc, M, Nn, Kd);
}

extern "C" void kernel_launch(void* const* d_in, const int* in_sizes, int n_in,
                              void* d_out, int out_size, void* d_ws, size_t ws_size,
                              hipStream_t stream)
{
  (void)in_sizes; (void)n_in; (void)out_size; (void)ws_size;

  // JAX tree-flatten order (sorted dict keys): enc, head, layers, pos, pts
  const float* eW1 = (const float*)d_in[0];   // [3,128]
  const float* eW2 = (const float*)d_in[1];   // [128,256]
  const float* eW3 = (const float*)d_in[2];   // [512,512]
  const float* eW4 = (const float*)d_in[3];   // [512,384]
  const float* eb1 = (const float*)d_in[4];
  const float* eb2 = (const float*)d_in[5];
  const float* eb3 = (const float*)d_in[6];
  const float* eb4 = (const float*)d_in[7];
  const float* ebe1 = (const float*)d_in[8];
  const float* ebe2 = (const float*)d_in[9];
  const float* ebe3 = (const float*)d_in[10];
  const float* ebe4 = (const float*)d_in[11];
  const float* eg1 = (const float*)d_in[12];
  const float* eg2 = (const float*)d_in[13];
  const float* eg3 = (const float*)d_in[14];
  const float* eg4 = (const float*)d_in[15];
  const float* hW1 = (const float*)d_in[16];  // [384,256]
  const float* hW2 = (const float*)d_in[17];  // [256,128]
  const float* hW3 = (const float*)d_in[18];  // [128,40]
  const float* hb1 = (const float*)d_in[19];
  const float* hb2 = (const float*)d_in[20];
  const float* hb3 = (const float*)d_in[21];
  const float* hbe1 = (const float*)d_in[22];
  const float* hbe2 = (const float*)d_in[23];
  const float* hg1 = (const float*)d_in[24];
  const float* hg2 = (const float*)d_in[25];
  const float* lAlog = (const float*)d_in[26]; // [12,768,16]
  const float* lDp   = (const float*)d_in[27]; // [12,768]
  const float* lWdt  = (const float*)d_in[28]; // [12,768,24]
  const float* lWin  = (const float*)d_in[29]; // [12,1536,384]
  const float* lWout = (const float*)d_in[30]; // [12,384,768]
  const float* lWx   = (const float*)d_in[31]; // [12,56,768]
  const float* lbdt  = (const float*)d_in[32]; // [12,768]
  const float* lbnb  = (const float*)d_in[33]; // [12,384]
  const float* lbng  = (const float*)d_in[34]; // [12,384]
  const float* lcvb  = (const float*)d_in[35]; // [12,768]
  const float* lcvw  = (const float*)d_in[36]; // [12,768,4]
  const float* pW1 = (const float*)d_in[37];   // [3,128]
  const float* pW2 = (const float*)d_in[38];   // [128,384]
  const float* pb1 = (const float*)d_in[39];
  const float* pb2 = (const float*)d_in[40];
  const float* pbe1 = (const float*)d_in[41];
  const float* pbe2 = (const float*)d_in[42];
  const float* pg1 = (const float*)d_in[43];
  const float* pg2 = (const float*)d_in[44];
  const float* pts = (const float*)d_in[45];   // [8,1024,3]

  // ---- workspace bump allocator ----
  size_t cur = 0;
  char* base = (char*)d_ws;
  auto alloc = [&](size_t bytes) -> char* {
    char* p = base + cur;
    cur = (cur + bytes + 255) & ~(size_t)255;
    return p;
  };
  int*       idx   = (int*)alloc((size_t)B_ * G_ * 4);
  float*     cen0  = (float*)alloc((size_t)B_ * G_ * 3 * 4);
  _Float16*  cenh  = (_Float16*)alloc((size_t)MTOK * 8 * 2);
  _Float16*  nb    = (_Float16*)alloc((size_t)MPT * 8 * 2);
  _Float16*  Y1    = (_Float16*)alloc((size_t)MPT * 128 * 2);
  _Float16*  Y2    = (_Float16*)alloc((size_t)MPT * 256 * 2);
  _Float16*  CAT   = (_Float16*)alloc((size_t)MPT * 512 * 2);
  _Float16*  Y3    = (_Float16*)alloc((size_t)MPT * 512 * 2);
  _Float16*  Y4    = (_Float16*)alloc((size_t)MPT * 384 * 2);
  _Float16*  henc  = (_Float16*)alloc((size_t)MTOK * D_ * 2);
  _Float16*  pos1  = (_Float16*)alloc((size_t)MTOK * 128 * 2);
  _Float16*  pose  = (_Float16*)alloc((size_t)MTOK * D_ * 2);
  _Float16*  hbuf  = (_Float16*)alloc((size_t)MTOK * D_ * 2);
  _Float16*  rbuf  = (_Float16*)alloc((size_t)MTOK * D_ * 2);
  _Float16*  hb    = (_Float16*)alloc((size_t)MTOK * D_ * 2);
  _Float16*  xz    = (_Float16*)alloc((size_t)MTOK * 2 * DI_ * 2);
  _Float16*  sx    = (_Float16*)alloc((size_t)MTOK * DI_ * 2);
  _Float16*  sz    = (_Float16*)alloc((size_t)MTOK * DI_ * 2);
  _Float16*  dbl   = (_Float16*)alloc((size_t)MTOK * 56 * 2);
  _Float16*  dtraw = (_Float16*)alloc((size_t)MTOK * DI_ * 2);
  float*     dtf   = (float*)alloc((size_t)MTOK * DI_ * 4);
  _Float16*  ybuf  = (_Float16*)alloc((size_t)MTOK * DI_ * 2);
  _Float16*  pool  = (_Float16*)alloc((size_t)TB_ * D_ * 2);
  _Float16*  s0    = (_Float16*)alloc((size_t)TB_ * D_ * 2);
  _Float16*  t1    = (_Float16*)alloc((size_t)TB_ * 256 * 2);
  _Float16*  t2    = (_Float16*)alloc((size_t)TB_ * 128 * 2);
  _Float16*  lg    = (_Float16*)alloc((size_t)TB_ * CLS_ * 2);
  float*     stsum = (float*)alloc(2048 * 4);
  float*     stss  = (float*)alloc(2048 * 4);
  // f16 [N][K] weights
  _Float16*  eW1t  = (_Float16*)alloc((size_t)128 * 8 * 2);
  _Float16*  eW2t  = (_Float16*)alloc((size_t)256 * 128 * 2);
  _Float16*  eW3t  = (_Float16*)alloc((size_t)512 * 512 * 2);
  _Float16*  eW4t  = (_Float16*)alloc((size_t)384 * 512 * 2);
  _Float16*  pW1t  = (_Float16*)alloc((size_t)128 * 8 * 2);
  _Float16*  pW2t  = (_Float16*)alloc((size_t)384 * 128 * 2);
  _Float16*  hW1t  = (_Float16*)alloc((size_t)256 * 384 * 2);
  _Float16*  hW2t  = (_Float16*)alloc((size_t)128 * 256 * 2);
  _Float16*  hW3t  = (_Float16*)alloc((size_t)40 * 128 * 2);
  _Float16*  WinT  = (_Float16*)alloc((size_t)DEPTH_ * 2 * DI_ * D_ * 2);
  _Float16*  WxT   = (_Float16*)alloc((size_t)DEPTH_ * 56 * DI_ * 2);
  _Float16*  WdtT  = (_Float16*)alloc((size_t)DEPTH_ * DI_ * DTR_ * 2);
  _Float16*  WoutT = (_Float16*)alloc((size_t)DEPTH_ * D_ * DI_ * 2);

  auto prep = [&](const float* W, _Float16* Wt, int Kd, int Nn, int ldw, int trans) {
    size_t tot = (size_t)Nn * ldw;
    k_wprep<<<CDIV(tot, 256), 256, 0, stream>>>(W, Wt, Kd, Nn, ldw, trans);
  };
  prep(eW1, eW1t, 3, 128, 8, 1);
  prep(eW2, eW2t, 128, 256, 128, 1);
  prep(eW3, eW3t, 512, 512, 512, 1);
  prep(eW4, eW4t, 512, 384, 512, 1);
  prep(pW1, pW1t, 3, 128, 8, 1);
  prep(pW2, pW2t, 128, 384, 128, 1);
  prep(hW1, hW1t, 384, 256, 384, 1);
  prep(hW2, hW2t, 256, 128, 256, 1);
  prep(hW3, hW3t, 128, 40, 128, 1);
  for (int l = 0; l < DEPTH_; ++l) {
    prep(lWin  + (size_t)l * 2 * DI_ * D_, WinT  + (size_t)l * 2 * DI_ * D_, D_,  2 * DI_, D_,  0);
    prep(lWx   + (size_t)l * 56 * DI_,     WxT   + (size_t)l * 56 * DI_,     DI_, 56,      DI_, 0);
    prep(lWdt  + (size_t)l * DI_ * DTR_,   WdtT  + (size_t)l * DI_ * DTR_,   DTR_, DI_,    DTR_, 0);
    prep(lWout + (size_t)l * D_ * DI_,     WoutT + (size_t)l * D_ * DI_,     DI_, D_,      DI_, 0);
  }

  auto zstats = [&]() {
    (void)hipMemsetAsync(stsum, 0, 2048 * 4, stream);
    (void)hipMemsetAsync(stss,  0, 2048 * 4, stream);
  };
  auto stats = [&](const _Float16* X, int C, int M) {
    zstats();
    dim3 g(CDIV(C, 256), M < 256 ? M : 256);
    k_stats<<<g, 256, 0, stream>>>(X, C, M, stsum, stss);
  };
  auto bnap = [&](const _Float16* X, int C, int M, _Float16* Y, int ldy, int yoff,
                  const float* gam, const float* bet) {
    k_bn_apply<<<CDIV((size_t)M * C, 256), 256, 0, stream>>>(
        X, C, M, Y, ldy, yoff, stsum, stss, gam, bet, (float)M);
  };
  auto lif = [&](const _Float16* X, int ldx, int xoff, _Float16* Y, int ldy,
                 int yoff, int C, int Mhalf, bool bn, const float* gam,
                 const float* bet, float cnt) {
    k_lif<<<CDIV((size_t)Mhalf * C, 256), 256, 0, stream>>>(
        X, ldx, xoff, Y, ldy, yoff, C, Mhalf,
        bn ? stsum : (const float*)nullptr, stss, gam, bet, cnt);
  };

  // ---- FPS + KNN + gather ----
  k_fps<<<B_, 256, 0, stream>>>(pts, idx, cen0, cenh);
  k_knn<<<dim3(G_, B_), 128, 0, stream>>>(pts, cen0, nb);

  // ---- encoder ----
  gemm(stream, nb, 8, eW1t, 8, eb1, Y1, 128, MPT, 128, 3);
  stats(Y1, 128, MPT);
  lif(Y1, 128, 0, Y1, 128, 0, 128, MPT / 2, true, eg1, ebe1, (float)MPT);
  gemm(stream, Y1, 128, eW2t, 128, eb2, Y2, 256, MPT, 256, 128);
  stats(Y2, 256, MPT);
  bnap(Y2, 256, MPT, CAT, 512, 256, eg2, ebe2);
  k_maxk<<<CDIV((size_t)MTOK * 256, 256), 256, 0, stream>>>(
      CAT, 512, 256, CAT, 512, 0, MTOK, K_, 256, 1);
  lif(CAT, 512, 0, CAT, 512, 0, 512, MPT / 2, false, nullptr, nullptr, 1.0f);
  gemm(stream, CAT, 512, eW3t, 512, eb3, Y3, 512, MPT, 512, 512);
  stats(Y3, 512, MPT);
  lif(Y3, 512, 0, Y3, 512, 0, 512, MPT / 2, true, eg3, ebe3, (float)MPT);
  gemm(stream, Y3, 512, eW4t, 512, eb4, Y4, ENC_, MPT, ENC_, 512);
  stats(Y4, ENC_, MPT);
  bnap(Y4, ENC_, MPT, Y4, ENC_, 0, eg4, ebe4);
  k_maxk<<<CDIV((size_t)MTOK * ENC_, 256), 256, 0, stream>>>(
      Y4, ENC_, 0, henc, ENC_, 0, MTOK, K_, ENC_, 0);

  // ---- positional embed ----
  gemm(stream, cenh, 8, pW1t, 8, pb1, pos1, 128, MTOK, 128, 3);
  stats(pos1, 128, MTOK);
  lif(pos1, 128, 0, pos1, 128, 0, 128, MTOK / 2, true, pg1, pbe1, (float)MTOK);
  gemm(stream, pos1, 128, pW2t, 128, pb2, pose, D_, MTOK, D_, 128);
  stats(pose, D_, MTOK);
  bnap(pose, D_, MTOK, pose, D_, 0, pg2, pbe2);

  size_t nh = (size_t)MTOK * D_;
  k_add<<<CDIV(nh, 256), 256, 0, stream>>>(henc, pose, hbuf, nh);
  (void)hipMemsetAsync(rbuf, 0, nh * 2, stream);

  // ---- mamba layers ----
  for (int l = 0; l < DEPTH_; ++l) {
    k_add<<<CDIV(nh, 256), 256, 0, stream>>>(hbuf, rbuf, rbuf, nh);
    stats(rbuf, D_, MTOK);
    bnap(rbuf, D_, MTOK, hb, D_, 0, lbng + l * D_, lbnb + l * D_);
    gemm(stream, hb, D_, WinT + (size_t)l * 2 * DI_ * D_, D_, nullptr,
         xz, 2 * DI_, MTOK, 2 * DI_, D_);
    k_conv_lif<<<CDIV((size_t)B_ * G_ * DI_, 256), 256, 0, stream>>>(
        xz, lcvw + (size_t)l * DI_ * DC_, lcvb + (size_t)l * DI_, sx);
    lif(xz, 2 * DI_, DI_, sz, DI_, 0, DI_, MTOK / 2, false, nullptr, nullptr, 1.0f);
    gemm(stream, sx, DI_, WxT + (size_t)l * 56 * DI_, DI_, nullptr,
         dbl, 56, MTOK, 56, DI_);
    gemm(stream, dbl, 56, WdtT + (size_t)l * DI_ * DTR_, DTR_, nullptr,
         dtraw, DI_, MTOK, DI_, DTR_);
    k_softplus<<<CDIV((size_t)MTOK * DI_, 256), 256, 0, stream>>>(
        dtraw, lbdt + (size_t)l * DI_, dtf, (size_t)MTOK * DI_);
    k_scan<<<CDIV(TB_ * DI_, 256), 256, 0, stream>>>(
        dtf, dbl, sx, sz, lAlog + (size_t)l * DI_ * DS_, lDp + (size_t)l * DI_, ybuf);
    gemm(stream, ybuf, DI_, WoutT + (size_t)l * D_ * DI_, DI_, nullptr,
         hbuf, D_, MTOK, D_, DI_);
  }
  k_add<<<CDIV(nh, 256), 256, 0, stream>>>(hbuf, rbuf, hbuf, nh);

  // ---- head ----
  k_maxk<<<CDIV((size_t)TB_ * D_, 256), 256, 0, stream>>>(
      hbuf, D_, 0, pool, D_, 0, TB_, G_, D_, 0);
  lif(pool, D_, 0, s0, D_, 0, D_, TB_ / 2, false, nullptr, nullptr, 1.0f);
  gemm(stream, s0, D_, hW1t, D_, hb1, t1, 256, TB_, 256, D_);
  stats(t1, 256, TB_);
  lif(t1, 256, 0, t1, 256, 0, 256, TB_ / 2, true, hg1, hbe1, (float)TB_);
  gemm(stream, t1, 256, hW2t, 256, hb2, t2, 128, TB_, 128, 256);
  stats(t2, 128, TB_);
  lif(t2, 128, 0, t2, 128, 0, 128, TB_ / 2, true, hg2, hbe2, (float)TB_);
  gemm(stream, t2, 128, hW3t, 128, hb3, lg, CLS_, TB_, CLS_, 128);
  k_final<<<CDIV(B_ * CLS_, 256), 256, 0, stream>>>(lg, (float*)d_out);
}
